// LowRank1d_44169443672694
// MI455X (gfx1250) — matrix operand order repaired and checked by reading, not compiled, not run on hardware
//
#include <hip/hip_runtime.h>
#include <hip/hip_bf16.h>

// MI455X / gfx1250, wave32. bf16 WMMA everywhere; async global->LDS staging.

typedef __bf16 bf16;
typedef __attribute__((ext_vector_type(16))) __bf16 v16bf;
typedef __attribute__((ext_vector_type(8)))  __bf16 v8bf;
typedef __attribute__((ext_vector_type(8)))  float  v8f;

#define WIDTH  128
#define RANK   4
#define NPTS   8192
#define BATCH  8

// ---- workspace layout (bytes) ----
// [0,   2MB)  : AM   f32  [B][RANK][128][128]  (atomic accumulators)
// [2MB, 3MB)  : AMT  bf16 [B][RANK][128 c][128 d]  (transposed, for qo staging)
// [3MB, 3MB+672KB) : bf16 weights [2 nets][W1T | W2T | W3T]
// [4MB, 4MB+16.8MB): vT  bf16 [B][128 c][8192 m]
#define AM_ELEMS   (BATCH * RANK * WIDTH * WIDTH)       // 524288 f32
#define AMT_OFF    (2u << 20)
#define WTS_OFF    (3u << 20)
#define VT_OFF     (4u << 20)
#define W1T_ELEMS  (128 * 64)
#define W2T_ELEMS  (256 * 128)
#define W3T_ELEMS  (4 * 128 * 256)
#define NET_ELEMS  (W1T_ELEMS + W2T_ELEMS + W3T_ELEMS)  // 172032 bf16

struct NetW {
  const float* W0;  const float* b0;   // f32 [2][64], [64]
  const bf16*  W1T; const float* b1;   // bf16 [128][64]
  const bf16*  W2T; const float* b2;   // bf16 [256][128]
  const bf16*  W3T; const float* b3;   // bf16 [4][128][256] per-head, b3 f32 [512]
};

// ---------------- async global->LDS (16B per lane, ASYNCcnt) ----------------
__device__ __forceinline__ uint32_t lds_off_of(const void* p) {
  // generic LDS pointer: addr[31:0] is the LDS byte offset (ISA 10.2 aperture rule)
  return (uint32_t)(uintptr_t)p;
}
__device__ __forceinline__ void async_copy16(uint32_t lds_byte_off, const void* gaddr) {
  asm volatile("global_load_async_to_lds_b128 %0, %1, off"
               :: "v"(lds_byte_off), "v"(gaddr) : "memory");
}
__device__ __forceinline__ void async_wait0() {
  asm volatile("s_wait_asynccnt 0x0" ::: "memory");
}

// ---------------- WMMA fragment builders (ISA 7.12.2 layouts) ----------------
// A fragment, source row-major S[M][K]; lane r = M row, contiguous K runs.
__device__ __forceinline__ v16bf fragA_row(const bf16* base, int ld, int lane) {
  int r = lane & 15, half = lane >> 4;
  const v8bf* p = (const v8bf*)(base + r * ld + half * 8);   // 16B aligned
  return __builtin_shufflevector(p[0], p[2], 0, 1, 2, 3, 4, 5, 6, 7,
                                 8, 9, 10, 11, 12, 13, 14, 15);
}
// B fragment from B^T stored [N][K]; lane r = N col, same K runs as fragA_row.
__device__ __forceinline__ v16bf fragB_rowT(const bf16* base, int ld, int lane) {
  int r = lane & 15, half = lane >> 4;
  const v8bf* p = (const v8bf*)(base + r * ld + half * 8);   // 16B aligned
  return __builtin_shufflevector(p[0], p[2], 0, 1, 2, 3, 4, 5, 6, 7,
                                 8, 9, 10, 11, 12, 13, 14, 15);
}
__device__ __forceinline__ v8f wmma_bf16(v16bf a, v16bf b, v8f c) {
  return __builtin_amdgcn_wmma_f32_16x16x32_bf16(false, a, false, b, (short)0, c,
                                                 false, false);
}

// ---------------- tiled MLP layer: OUT = act(IN[32][K] @ W + bias) ----------------
template <int K, int N, bool RELU, bool STORE_T>
__device__ __forceinline__ void gemm_layer(const bf16* IN, const bf16* WT,
                                           const float* bias, int bstride, int boff,
                                           bf16* OUT, int lane, int wave) {
  constexpr int MT = 2, NT = N / 16, TILES = MT * NT;
  int half = lane >> 4, col = lane & 15;
  for (int t = wave; t < TILES; t += 8) {
    int tm = t % MT, tn = t / MT;
    v8f c = {};
#pragma unroll
    for (int k = 0; k < K; k += 32) {
      v16bf af = fragA_row(IN + tm * 16 * K + k, K, lane);
      v16bf bf = fragB_rowT(WT + tn * 16 * K + k, K, lane);
      c = wmma_bf16(af, bf, c);
    }
    float bv = bias[(tn * 16 + col) * bstride + boff];
    if (STORE_T) {  // OUT[N][32] transposed -> one contiguous v8bf store
      v8bf o;
#pragma unroll
      for (int r = 0; r < 8; ++r) {
        float y = c[r] + bv;
        if (RELU) y = fmaxf(y, 0.f);
        o[r] = (bf16)y;
      }
      *(v8bf*)(OUT + (tn * 16 + col) * 32 + tm * 16 + half * 8) = o;
    } else {        // OUT[32][N] row-major
#pragma unroll
      for (int r = 0; r < 8; ++r) {
        float y = c[r] + bv;
        if (RELU) y = fmaxf(y, 0.f);
        OUT[(tm * 16 + half * 8 + r) * N + tn * 16 + col] = (bf16)y;
      }
    }
  }
}

// layers 0..2 for 32 points -> sm_h3 [32][256] bf16.
__device__ __forceinline__ void mlp_trunk(const float* a_ptr, const NetW& nw,
                                          bf16* sm_h1, bf16* sm_h2, bf16* sm_h3,
                                          int lane, int wave, int tid) {
  for (int j = tid; j < 32 * 64; j += 256) {   // layer0 (K=2) on VALU
    int m = j >> 6, n = j & 63;
    float y = fmaf(a_ptr[m * 2 + 0], nw.W0[n],
             fmaf(a_ptr[m * 2 + 1], nw.W0[64 + n], nw.b0[n]));
    sm_h1[m * 64 + n] = (bf16)fmaxf(y, 0.f);
  }
  __syncthreads();
  gemm_layer<64, 128, true, false>(sm_h1, nw.W1T, nw.b1, 1, 0, sm_h2, lane, wave);
  __syncthreads();
  gemm_layer<128, 256, true, false>(sm_h2, nw.W2T, nw.b2, 1, 0, sm_h3, lane, wave);
  __syncthreads();
}

// ------------- kernel 0: weight prep + v -> vT bf16 + zero AM -------------
__global__ __launch_bounds__(256) void prep_kernel(
    const float* phiW1, const float* phiW2, const float* phiW3,
    const float* psiW1, const float* psiW2, const float* psiW3,
    const float* v, bf16* wts, bf16* vT, float* AM) {
  int i = blockIdx.x * blockDim.x + threadIdx.x;
  int stride = gridDim.x * blockDim.x;
  for (int j = i; j < AM_ELEMS; j += stride) AM[j] = 0.f;
  // vT[b][c][m] = v[b][m][c] (write-coalesced)
  for (long j = i; j < (long)BATCH * WIDTH * NPTS; j += stride) {
    long m = j & (NPTS - 1);
    long c = (j >> 13) & (WIDTH - 1);
    long b = j >> 20;
    vT[j] = (bf16)v[(b * NPTS + m) * WIDTH + c];
  }
  for (int net = 0; net < 2; ++net) {
    const float* W1 = net ? psiW1 : phiW1;
    const float* W2 = net ? psiW2 : phiW2;
    const float* W3 = net ? psiW3 : phiW3;
    bf16* d1 = wts + net * NET_ELEMS;
    bf16* d2 = d1 + W1T_ELEMS;
    bf16* d3 = d2 + W2T_ELEMS;
    for (int j = i; j < W1T_ELEMS; j += stride) {   // W1T[n][k] = W1[k][n]
      int n = j / 64, k = j % 64;
      d1[j] = (bf16)W1[k * 128 + n];
    }
    for (int j = i; j < W2T_ELEMS; j += stride) {   // W2T[n][k] = W2[k][n]
      int n = j / 128, k = j % 128;
      d2[j] = (bf16)W2[k * 256 + n];
    }
    for (int j = i; j < W3T_ELEMS; j += stride) {   // W3T[h][d][k] = W3[k][d*4+h]
      int h = j / (128 * 256), rem = j % (128 * 256);
      int d = rem / 256, k = rem % 256;
      d3[j] = (bf16)W3[k * 512 + d * RANK + h];
    }
  }
}

// ------------- kernel 1: attn_map[b][h] += K^T V (fused psi MLP) -------------
// grid (32, B, RANK); 256 threads = 8 waves; 8 subchunks of 32 points each.
__global__ __launch_bounds__(256) void kv_kernel(const float* __restrict__ a,
                                                 const bf16* __restrict__ vT,
                                                 NetW psi, float* __restrict__ AM) {
  __shared__ __attribute__((aligned(16))) bf16 sm_h3[32 * 256]; // 16 KB
  __shared__ __attribute__((aligned(16))) bf16 sm_scr[6144];    // 12 KB: h1|h2, then psiT[128][32]
  __shared__ __attribute__((aligned(16))) bf16 sm_vT[128 * 32]; //  8 KB: v^T [c][m]
  int b = blockIdx.y, h = blockIdx.z;
  int tid = threadIdx.x, lane = tid & 31, wave = tid >> 5;
  int half = lane >> 4, col = lane & 15;
  uint32_t vbase = lds_off_of(sm_vT);

  v8f acc[8];
#pragma unroll
  for (int t = 0; t < 8; ++t) acc[t] = (v8f){};

  int m_base = blockIdx.x * 256;
  for (int s = 0; s < 8; ++s) {
    int m0 = m_base + s * 32;
    // async-stage v^T chunk: rows of 64B (4 x 16B), 512 transfers / 256 threads
    {
      const bf16* vblk = vT + (size_t)b * WIDTH * NPTS + m0;
#pragma unroll
      for (int t = 0; t < 2; ++t) {
        int idx = t * 256 + tid;
        int c = idx >> 2, seg = idx & 3;
        async_copy16(vbase + (uint32_t)(c * 32 + seg * 8) * 2,
                     vblk + (size_t)c * NPTS + seg * 8);
      }
    }
    // overlap the DMA with the psi MLP trunk + head layer
    mlp_trunk(a + ((size_t)b * NPTS + m0) * 2, psi, sm_scr, sm_scr + 2048, sm_h3,
              lane, wave, tid);
    gemm_layer<256, 128, false, true>(sm_h3, psi.W3T + h * (128 * 256), psi.b3, RANK, h,
                                      sm_scr, lane, wave);
    async_wait0();
    __syncthreads();
    // D[d][c] += psi^T (d x m) * v (m x c); K=32. wave owns d-tile=wave, all 8 c-tiles.
    v16bf af = fragA_row(sm_scr + wave * 16 * 32, 32, lane);
    v16bf bfr[8];
#pragma unroll
    for (int tc = 0; tc < 8; ++tc)
      bfr[tc] = fragB_rowT(sm_vT + tc * 16 * 32, 32, lane);
#pragma unroll
    for (int tc = 0; tc < 8; ++tc)
      acc[tc] = wmma_bf16(af, bfr[tc], acc[tc]);
    __syncthreads();
  }
  float* amh = AM + ((size_t)(b * RANK + h)) * WIDTH * WIDTH;
#pragma unroll
  for (int tc = 0; tc < 8; ++tc)
#pragma unroll
    for (int r = 0; r < 8; ++r) {
      int d = wave * 16 + half * 8 + r;
      unsafeAtomicAdd(&amh[d * WIDTH + tc * 16 + col], acc[tc][r]);
    }
}

// ------------- kernel 1.5: AMT[b][h][c][d] = bf16(AM[b][h][d][c]) -------------
__global__ __launch_bounds__(256) void am_compress(const float* __restrict__ AM,
                                                   bf16* __restrict__ AMT) {
  int i = blockIdx.x * blockDim.x + threadIdx.x;
  int stride = gridDim.x * blockDim.x;
  for (int j = i; j < AM_ELEMS; j += stride) {
    int d = j & 127, c = (j >> 7) & 127, bh = j >> 14;
    AMT[j] = (bf16)AM[bh * 16384 + d * 128 + c];
  }
}

// ------- kernel 2: out[b] = (sum_h Q_h @ attn_map[b][h]) / n (fused phi MLP) -------
__global__ __launch_bounds__(256) void qo_kernel(const float* __restrict__ a,
                                                 NetW phi,
                                                 const bf16* __restrict__ AMT,
                                                 float* __restrict__ out) {
  __shared__ __attribute__((aligned(16))) bf16 sm_h3[32 * 256];        // 16 KB
  __shared__ __attribute__((aligned(16))) bf16 sm_scr[6144];           // 12 KB: h1|h2, then phi_h
  __shared__ __attribute__((aligned(16))) bf16 sm_amT[WIDTH * WIDTH];  // 32 KB: AM^T [c][d]
  int b = blockIdx.y, m0 = blockIdx.x * 32;
  int tid = threadIdx.x, lane = tid & 31, wave = tid >> 5;
  int half = lane >> 4, col = lane & 15;
  uint32_t abase = lds_off_of(sm_amT);

  v8f acc[2];
  acc[0] = (v8f){}; acc[1] = (v8f){};

  mlp_trunk(a + ((size_t)b * NPTS + m0) * 2, phi, sm_scr, sm_scr + 2048, sm_h3,
            lane, wave, tid);
  for (int h = 0; h < RANK; ++h) {
    // async-stage AM^T block (32KB contiguous: 2048 x 16B), overlap with layer3
    {
      const bf16* blk = AMT + (size_t)(b * RANK + h) * (WIDTH * WIDTH);
#pragma unroll
      for (int t = 0; t < 8; ++t) {
        int idx = t * 256 + tid;
        async_copy16(abase + (uint32_t)idx * 16, (const char*)blk + (size_t)idx * 16);
      }
    }
    gemm_layer<256, 128, false, false>(sm_h3, phi.W3T + h * (128 * 256), phi.b3, RANK, h,
                                       sm_scr, lane, wave);
    async_wait0();
    __syncthreads();
    // out_tile[m][c] += phi_h (m x d) * am (d x c); 16 tiles / 8 waves, pipelined frags
#pragma unroll
    for (int j = 0; j < 2; ++j) {
      int t = wave + 8 * j, tm = t % 2, tc = t / 2;
      v16bf afs[4], bfs[4];
#pragma unroll
      for (int kk = 0; kk < 4; ++kk) {
        afs[kk] = fragA_row(sm_scr + tm * 16 * 128 + kk * 32, 128, lane);
        bfs[kk] = fragB_rowT(sm_amT + tc * 16 * 128 + kk * 32, 128, lane);
      }
#pragma unroll
      for (int kk = 0; kk < 4; ++kk)
        acc[j] = wmma_bf16(afs[kk], bfs[kk], acc[j]);
    }
    __syncthreads();
  }
  const float inv_n = 1.0f / (float)NPTS;
#pragma unroll
  for (int j = 0; j < 2; ++j) {
    int t = wave + 8 * j, tm = t % 2, tc = t / 2;
#pragma unroll
    for (int r = 0; r < 8; ++r) {
      int m = m0 + tm * 16 + half * 8 + r;
      out[((size_t)b * NPTS + m) * WIDTH + tc * 16 + col] = acc[j][r] * inv_n;
    }
  }
}

extern "C" void kernel_launch(void* const* d_in, const int* in_sizes, int n_in,
                              void* d_out, int out_size, void* d_ws, size_t ws_size,
                              hipStream_t stream) {
  (void)in_sizes; (void)n_in; (void)out_size; (void)ws_size;
  const float* v = (const float*)d_in[0];
  const float* a = (const float*)d_in[1];
  const float* phiW0 = (const float*)d_in[2];  const float* phib0 = (const float*)d_in[3];
  const float* phiW1 = (const float*)d_in[4];  const float* phib1 = (const float*)d_in[5];
  const float* phiW2 = (const float*)d_in[6];  const float* phib2 = (const float*)d_in[7];
  const float* phiW3 = (const float*)d_in[8];  const float* phib3 = (const float*)d_in[9];
  const float* psiW0 = (const float*)d_in[10]; const float* psib0 = (const float*)d_in[11];
  const float* psiW1 = (const float*)d_in[12]; const float* psib1 = (const float*)d_in[13];
  const float* psiW2 = (const float*)d_in[14]; const float* psib2 = (const float*)d_in[15];
  const float* psiW3 = (const float*)d_in[16]; const float* psib3 = (const float*)d_in[17];

  float* AM  = (float*)d_ws;
  bf16*  AMT = (bf16*)((char*)d_ws + AMT_OFF);
  bf16*  wts = (bf16*)((char*)d_ws + WTS_OFF);
  bf16*  vT  = (bf16*)((char*)d_ws + VT_OFF);

  prep_kernel<<<1024, 256, 0, stream>>>(phiW1, phiW2, phiW3, psiW1, psiW2, psiW3,
                                        v, wts, vT, AM);

  NetW phiN, psiN;
  phiN.W0 = phiW0; phiN.b0 = phib0;
  phiN.W1T = wts;                         phiN.b1 = phib1;
  phiN.W2T = wts + W1T_ELEMS;             phiN.b2 = phib2;
  phiN.W3T = wts + W1T_ELEMS + W2T_ELEMS; phiN.b3 = phib3;
  psiN.W0 = psiW0; psiN.b0 = psib0;
  psiN.W1T = wts + NET_ELEMS;                         psiN.b1 = psib1;
  psiN.W2T = wts + NET_ELEMS + W1T_ELEMS;             psiN.b2 = psib2;
  psiN.W3T = wts + NET_ELEMS + W1T_ELEMS + W2T_ELEMS; psiN.b3 = psib3;

  kv_kernel<<<dim3(NPTS / 256, BATCH, RANK), 256, 0, stream>>>(a, vT, psiN, AM);
  am_compress<<<512, 256, 0, stream>>>(AM, AMT);
  qo_kernel<<<dim3(NPTS / 32, BATCH), 256, 0, stream>>>(a, phiN, AMT, (float*)d_out);
}
